// OIMLoss_80083960201221
// MI455X (gfx1250) — matrix operand order repaired and checked by reading, not compiled, not run on hardware
//
#include <hip/hip_runtime.h>
#include <hip/hip_bf16.h>
#include <stdint.h>

// C[2048, 10788] = 30 * A[2048,256] @ B[10788,256]^T
//   A = inputs, B = [lut(5532); cq(5000); non_id(256)]
// Strategy: fp32 -> (bf16 hi, bf16 lo) split planes in d_ws (one pass),
// then register-blocked WMMA GEMM: workgroup tile 128x128 (8 waves),
// wave tile 64(M) x 32(N) = 8 accumulator tiles, K stepped by 32 with
// v_wmma_f32_16x16x32_bf16, 3 products (hi*hi + hi*lo + lo*hi) per tile
// for near-fp32 accuracy. Operands (13 MB) live in the 192 MB L2, so we
// read fragments straight from global (L2 hits), no LDS staging needed.
//
// Workspace requirement: 13,238,272 bytes (~12.7 MiB).

typedef __attribute__((ext_vector_type(16))) __bf16 v16bf;
typedef __attribute__((ext_vector_type(8)))  float  v8f;

#define KDIM 256
#define MTOT 2048
#define NTOT 10788
#define NPAD 10880          // 85 * 128
#define OIM_SCALE 30.0f

// ---------------- fp32 -> bf16 hi/lo split ----------------
__global__ void split_bf16_kernel(const float* __restrict__ src,
                                  __bf16* __restrict__ dh,
                                  __bf16* __restrict__ dl,
                                  int n)
{
    int i = (blockIdx.x * blockDim.x + threadIdx.x) * 4;
    if (i >= n) return;
    float4 x = *(const float4*)(src + i);
    float xs[4] = {x.x, x.y, x.z, x.w};
    union { __bf16 h[4]; uint64_t u; } ph, pl;
#pragma unroll
    for (int j = 0; j < 4; ++j) {
        __bf16 hi = (__bf16)xs[j];                 // RNE truncation to bf16
        float  r  = xs[j] - (float)hi;             // residual
        ph.h[j] = hi;
        pl.h[j] = (__bf16)r;
    }
    *(uint64_t*)(dh + i) = ph.u;
    *(uint64_t*)(dl + i) = pl.u;
}

// ---------------- WMMA GEMM ----------------
union FragB16 { v16bf v; uint4 q[2]; };

__global__ __launch_bounds__(256)
void oim_gemm_kernel(const __bf16* __restrict__ Ah, const __bf16* __restrict__ Al,
                     const __bf16* __restrict__ Bh, const __bf16* __restrict__ Bl,
                     float* __restrict__ out)
{
    const int lane = threadIdx.x & 31;
    const int wave = threadIdx.x >> 5;
    const int wm   = wave >> 2;                    // 0..1
    const int wn   = wave & 3;                     // 0..3
    const int m0   = blockIdx.y * 128 + wm * 64;   // wave M origin (4 tiles of 16)
    const int n0   = blockIdx.x * 128 + wn * 32;   // wave N origin (2 tiles of 16)

    const int lrow = lane & 15;
    const int lhi  = lane >> 4;                    // 0 or 1 (upper half-wave)

    v8f acc[4][2];
#pragma unroll
    for (int mi = 0; mi < 4; ++mi)
#pragma unroll
        for (int ni = 0; ni < 2; ++ni) {
            v8f z = {0.f, 0.f, 0.f, 0.f, 0.f, 0.f, 0.f, 0.f};
            acc[mi][ni] = z;
        }

    // A fragment (16-bit A 16x32): lane holds row m0+mi*16+lrow,
    //   K elements [kb, kb+8) and [kb+16, kb+24), kb = k0 + lhi*8
    // B fragment (16-bit B 32x16): lane holds weight row n0+ni*16+lrow,
    //   K elements [kb, kb+8) and [kb+16, kb+24), kb = k0 + lhi*8
    const __bf16* aH = Ah + (size_t)(m0 + lrow) * KDIM + lhi * 8;
    const __bf16* aL = Al + (size_t)(m0 + lrow) * KDIM + lhi * 8;
    const __bf16* bH = Bh + (size_t)(n0 + lrow) * KDIM + lhi * 8;
    const __bf16* bL = Bl + (size_t)(n0 + lrow) * KDIM + lhi * 8;

    for (int k0 = 0; k0 < KDIM; k0 += 32) {
        FragB16 ah[4], al[4], bh[2], bl[2];
#pragma unroll
        for (int mi = 0; mi < 4; ++mi) {
            const __bf16* p  = aH + (size_t)mi * 16 * KDIM + k0;
            ah[mi].q[0] = *(const uint4*)(p);
            ah[mi].q[1] = *(const uint4*)(p + 16);
            const __bf16* pl = aL + (size_t)mi * 16 * KDIM + k0;
            al[mi].q[0] = *(const uint4*)(pl);
            al[mi].q[1] = *(const uint4*)(pl + 16);
        }
#pragma unroll
        for (int ni = 0; ni < 2; ++ni) {
            const __bf16* p  = bH + (size_t)ni * 16 * KDIM + k0;
            bh[ni].q[0] = *(const uint4*)(p);
            bh[ni].q[1] = *(const uint4*)(p + 16);
            const __bf16* pl = bL + (size_t)ni * 16 * KDIM + k0;
            bl[ni].q[0] = *(const uint4*)(pl);
            bl[ni].q[1] = *(const uint4*)(pl + 16);
        }
#pragma unroll
        for (int mi = 0; mi < 4; ++mi)
#pragma unroll
            for (int ni = 0; ni < 2; ++ni) {
                v8f c = acc[mi][ni];
                c = __builtin_amdgcn_wmma_f32_16x16x32_bf16(
                        false, ah[mi].v, false, bh[ni].v, (short)0, c, false, false);
                c = __builtin_amdgcn_wmma_f32_16x16x32_bf16(
                        false, ah[mi].v, false, bl[ni].v, (short)0, c, false, false);
                c = __builtin_amdgcn_wmma_f32_16x16x32_bf16(
                        false, al[mi].v, false, bh[ni].v, (short)0, c, false, false);
                acc[mi][ni] = c;
            }
    }

    // C/D layout: VGPR r -> M = r + 8*lhi (within 16-row tile), N = lrow
#pragma unroll
    for (int mi = 0; mi < 4; ++mi) {
#pragma unroll
        for (int ni = 0; ni < 2; ++ni) {
            int col = n0 + ni * 16 + lrow;
            if (col < NTOT) {
                float* o = out + (size_t)(m0 + mi * 16 + lhi * 8) * NTOT + col;
#pragma unroll
                for (int r = 0; r < 8; ++r)
                    o[(size_t)r * NTOT] = acc[mi][ni][r] * OIM_SCALE;
            }
        }
    }
}

extern "C" void kernel_launch(void* const* d_in, const int* in_sizes, int n_in,
                              void* d_out, int out_size, void* d_ws, size_t ws_size,
                              hipStream_t stream)
{
    (void)in_sizes; (void)n_in; (void)out_size; (void)ws_size;

    const float* inputs = (const float*)d_in[0];   // [2048, 256]
    const float* nonid  = (const float*)d_in[1];   // [256, 256]
    const float* lut    = (const float*)d_in[2];   // [5532, 256]
    const float* cq     = (const float*)d_in[3];   // [5000, 256]
    float* out = (float*)d_out;                    // [2048, 10788]

    // Workspace layout (bf16 planes)
    __bf16* Ah = (__bf16*)d_ws;                    // 2048*256
    __bf16* Al = Ah + (size_t)MTOT * KDIM;         // 2048*256
    __bf16* Bh = Al + (size_t)MTOT * KDIM;         // NPAD*256
    __bf16* Bl = Bh + (size_t)NPAD * KDIM;         // NPAD*256

    auto launch_split = [&](const float* s, __bf16* dh, __bf16* dl, int n) {
        int threads = 256;
        int blocks = (n / 4 + threads - 1) / threads;
        split_bf16_kernel<<<blocks, threads, 0, stream>>>(s, dh, dl, n);
    };

    // Build split planes: A, then B = [lut; cq; non_id] with zero pad to NPAD rows
    launch_split(inputs, Ah, Al, MTOT * KDIM);
    launch_split(lut,   Bh,                        Bl,                        5532 * KDIM);
    launch_split(cq,    Bh + (size_t)5532 * KDIM,  Bl + (size_t)5532 * KDIM,  5000 * KDIM);
    launch_split(nonid, Bh + (size_t)10532 * KDIM, Bl + (size_t)10532 * KDIM, 256 * KDIM);

    // Zero the pad rows [10788, 10880) so GEMM loads never go OOB
    size_t padBytes = (size_t)(NPAD - NTOT) * KDIM * sizeof(__bf16);
    hipMemsetAsync(Bh + (size_t)NTOT * KDIM, 0, padBytes, stream);
    hipMemsetAsync(Bl + (size_t)NTOT * KDIM, 0, padBytes, stream);

    // GEMM: workgroup tile 128(M) x 128(N); grid = 16 x 85
    dim3 grid(NPAD / 128, MTOT / 128);
    oim_gemm_kernel<<<dim3(grid.x, grid.y), 256, 0, stream>>>(Ah, Al, Bh, Bl, out);
}
